// SimpleCrossAttention_41248865911470
// MI455X (gfx1250) — matrix operand order repaired and checked by reading, not compiled, not run on hardware
//
#include <hip/hip_runtime.h>

typedef __attribute__((ext_vector_type(16))) _Float16 v16h;
typedef __attribute__((ext_vector_type(8)))  _Float16 v8h;
typedef __attribute__((ext_vector_type(8)))  float    v8f;

#define SEQ   2048
#define CDIM  512
#define HEADS 8
#define DHEAD 64

static __device__ __forceinline__ v16h pack16(v8h lo, v8h hi) {
  return __builtin_shufflevector(lo, hi, 0,1,2,3,4,5,6,7,8,9,10,11,12,13,14,15);
}

// ---------------------------------------------------------------------------
// LayerNorm (f32 in) -> f16 out.  One block per row, 128 threads, C=512.
// ---------------------------------------------------------------------------
__global__ void __launch_bounds__(128) ln_f16_kernel(
    const float* __restrict__ x, const float* __restrict__ g,
    const float* __restrict__ beta, _Float16* __restrict__ out, int C)
{
  int row = blockIdx.x;
  const float* xr = x + (size_t)row * C;
  float s = 0.f, ss = 0.f;
  for (int i = threadIdx.x; i < C; i += blockDim.x) {
    float v = xr[i]; s += v; ss += v * v;
  }
  #pragma unroll
  for (int m = 1; m < 32; m <<= 1) {
    s  += __shfl_xor(s,  m, 32);
    ss += __shfl_xor(ss, m, 32);
  }
  __shared__ float rs[4], rss[4];
  int w = threadIdx.x >> 5, ln = threadIdx.x & 31;
  if (ln == 0) { rs[w] = s; rss[w] = ss; }
  __syncthreads();
  if (threadIdx.x == 0) {
    float a = 0.f, b2 = 0.f;
    for (int i = 0; i < 4; ++i) { a += rs[i]; b2 += rss[i]; }
    rs[0] = a; rss[0] = b2;
  }
  __syncthreads();
  float mean = rs[0] / (float)C;
  float var  = rss[0] / (float)C - mean * mean;
  float rstd = rsqrtf(var + 1e-5f);
  _Float16* orow = out + (size_t)row * C;
  for (int i = threadIdx.x; i < C; i += blockDim.x) {
    float v = (xr[i] - mean) * rstd * g[i] + beta[i];
    orow[i] = (_Float16)v;
  }
}

// ---------------------------------------------------------------------------
// Weight convert+transpose: W[k][n] f32 -> Wt[n][k] f16.  grid.y = matrix id.
// ---------------------------------------------------------------------------
__global__ void __launch_bounds__(256) wconv_kernel(
    const float* __restrict__ W0, const float* __restrict__ W1,
    const float* __restrict__ W2, const float* __restrict__ W3,
    _Float16* __restrict__ Wt)
{
  int idx = blockIdx.x * 256 + threadIdx.x;          // 0 .. 512*512-1
  int k = idx >> 9, n = idx & 511;
  const float* W = (blockIdx.y == 0) ? W0 : (blockIdx.y == 1) ? W1
                 : (blockIdx.y == 2) ? W2 : W3;
  Wt[(size_t)blockIdx.y * CDIM * CDIM + (size_t)n * CDIM + k] =
      (_Float16)W[(size_t)k * CDIM + n];
}

// ---------------------------------------------------------------------------
// QKV projections.  C[2048x512] = A[2048x512] * W[512x512] + bias.
// grid = (8, 8, 12) : x = 256-row tile, y = 64-col tile, z = b*3 + proj.
// block = 256 (8 waves); each wave computes a 32x64 tile:
//   2 A-fragments x 4 B-fragments -> 8 WMMAs per k-step (B reused twice).
// Q,K stored [b,h,n,d]; V stored transposed [b,h,d,m].
// ---------------------------------------------------------------------------
__global__ void __launch_bounds__(256) qkv_gemm_kernel(
    const _Float16* __restrict__ xn, const _Float16* __restrict__ cn,
    const _Float16* __restrict__ Wt,
    const float* __restrict__ bq, const float* __restrict__ bk,
    const float* __restrict__ bv,
    _Float16* __restrict__ Q, _Float16* __restrict__ K,
    _Float16* __restrict__ Vt)
{
  int w    = threadIdx.x >> 5;
  int lane = threadIdx.x & 31;
  int m16  = lane & 15;
  int hi   = lane >> 4;
  int proj = blockIdx.z % 3;
  int b    = blockIdx.z / 3;
  int row0 = blockIdx.x * 256 + w * 32;
  int col0 = blockIdx.y * 64;

  const _Float16* A  = ((proj == 0) ? xn : cn) + (size_t)b * SEQ * CDIM;
  const _Float16* Bm = Wt + (size_t)proj * CDIM * CDIM;

  v8f acc[2][4] = {};
  const _Float16* arow0 = A + (size_t)(row0 + m16) * CDIM;
  const _Float16* arow1 = A + (size_t)(row0 + 16 + m16) * CDIM;

  for (int k = 0; k < CDIM; k += 32) {
    int k0 = k + hi * 8;   // A-frag: lanes0-15 K 0..7/16..23, lanes16-31 K 8..15/24..31
    v16h a0 = pack16(*(const v8h*)(arow0 + k0), *(const v8h*)(arow0 + k0 + 16));
    v16h a1 = pack16(*(const v8h*)(arow1 + k0), *(const v8h*)(arow1 + k0 + 16));
    #pragma unroll
    for (int t = 0; t < 4; ++t) {
      const _Float16* bp = Bm + (size_t)(col0 + t * 16 + m16) * CDIM + k + hi * 8;
      v16h bfrag = pack16(*(const v8h*)bp, *(const v8h*)(bp + 16));
      acc[0][t] = __builtin_amdgcn_wmma_f32_16x16x32_f16(
          false, a0, false, bfrag, (short)0, acc[0][t], false, false);
      acc[1][t] = __builtin_amdgcn_wmma_f32_16x16x32_f16(
          false, a1, false, bfrag, (short)0, acc[1][t], false, false);
    }
  }

  const float* bias = (proj == 0) ? bq : (proj == 1) ? bk : bv;
  #pragma unroll
  for (int g = 0; g < 2; ++g) {
    #pragma unroll
    for (int t = 0; t < 4; ++t) {
      int c  = col0 + t * 16 + m16;
      float bval = bias[c];
      int h = c >> 6, d = c & 63;
      #pragma unroll
      for (int r = 0; r < 8; ++r) {
        int rw = row0 + g * 16 + r + 8 * hi;
        _Float16 hv = (_Float16)(acc[g][t][r] + bval);
        if (proj == 0)
          Q[(((size_t)b * HEADS + h) * SEQ + rw) * DHEAD + d] = hv;
        else if (proj == 1)
          K[(((size_t)b * HEADS + h) * SEQ + rw) * DHEAD + d] = hv;
        else
          Vt[(((size_t)b * HEADS + h) * DHEAD + d) * SEQ + rw] = hv;
      }
    }
  }
}

// ---------------------------------------------------------------------------
// Flash attention.  grid = (32, 32): x = 64-query tile, y = b*H + h.
// block = 128 (4 waves); wave w owns query rows [q0+16w, q0+16w+16).
// K/V tiles staged once per block through LDS (padded rows, ds_load_b128
// fragment reloads); online softmax in f32; P staged through LDS to reshape
// f32 accumulators into f16 A-fragments.
// ---------------------------------------------------------------------------
__global__ void __launch_bounds__(128) attn_kernel(
    const _Float16* __restrict__ Q, const _Float16* __restrict__ K,
    const _Float16* __restrict__ Vt, _Float16* __restrict__ Aout)
{
  // kl: 64x(64+8) K tile | vl: 64x(64+8) V tile | pl: per-wave 16x(64+8) P
  __shared__ __align__(16) _Float16 lds[2 * 64 * 72 + 4 * 16 * 72];
  _Float16* kl = lds;
  _Float16* vl = lds + 64 * 72;

  int w    = threadIdx.x >> 5;
  int lane = threadIdx.x & 31;
  int m16  = lane & 15;
  int hi   = lane >> 4;
  int bh   = blockIdx.y;
  int b    = bh >> 3, h = bh & 7;
  int q0   = blockIdx.x * 64 + w * 16;

  const _Float16* Qp = Q  + ((size_t)bh * SEQ + q0) * DHEAD;
  const _Float16* Kp = K  + (size_t)bh * SEQ * DHEAD;
  const _Float16* Vp = Vt + (size_t)bh * DHEAD * SEQ;

  // Q A-fragments for this wave's 16 rows (d = 0..31 and 32..63)
  const _Float16* qr = Qp + (size_t)m16 * DHEAD + hi * 16;
  v16h qf0 = pack16(*(const v8h*)qr,        *(const v8h*)(qr + 8));
  v16h qf1 = pack16(*(const v8h*)(qr + 32), *(const v8h*)(qr + 40));

  float mrow[8], lrow[8];
  #pragma unroll
  for (int r = 0; r < 8; ++r) { mrow[r] = -1e30f; lrow[r] = 0.f; }
  v8f o[4] = {};

  _Float16* pl = lds + 2 * 64 * 72 + w * 16 * 72;
  const float scale = 0.125f;                 // 1/sqrt(64)

  for (int kb = 0; kb < SEQ; kb += 64) {
    // ---- stage K (64 keys x 64 d) and V^T (64 d x 64 keys) tiles to LDS ----
    #pragma unroll
    for (int i = 0; i < 4; ++i) {
      int c   = i * 128 + threadIdx.x;        // 0..511 chunk of 8 halves
      int row = c >> 3, off = (c & 7) * 8;
      *(v8h*)(kl + row * 72 + off) =
          *(const v8h*)(Kp + (size_t)(kb + row) * DHEAD + off);
      *(v8h*)(vl + row * 72 + off) =
          *(const v8h*)(Vp + (size_t)row * SEQ + kb + off);
    }
    __syncthreads();

    // ---- S = Q * K^T  (16 rows x 64 keys), K fragments from LDS ----
    v8f s[4] = {};
    #pragma unroll
    for (int t = 0; t < 4; ++t) {
      const _Float16* kr = kl + (size_t)(t * 16 + m16) * 72;
      v16h bf0 = pack16(*(const v8h*)(kr + hi * 16),
                        *(const v8h*)(kr + hi * 16 + 8));
      s[t] = __builtin_amdgcn_wmma_f32_16x16x32_f16(
          false, qf0, false, bf0, (short)0, s[t], false, false);
      v16h bf1 = pack16(*(const v8h*)(kr + 32 + hi * 16),
                        *(const v8h*)(kr + 32 + hi * 16 + 8));
      s[t] = __builtin_amdgcn_wmma_f32_16x16x32_f16(
          false, qf1, false, bf1, (short)0, s[t], false, false);
    }
    #pragma unroll
    for (int t = 0; t < 4; ++t)
      #pragma unroll
      for (int r = 0; r < 8; ++r) s[t][r] *= scale;

    // ---- online softmax: row max over 64 keys (half-wave reduction) ----
    #pragma unroll
    for (int r = 0; r < 8; ++r) {
      float v = fmaxf(fmaxf(s[0][r], s[1][r]), fmaxf(s[2][r], s[3][r]));
      #pragma unroll
      for (int m = 1; m < 16; m <<= 1) v = fmaxf(v, __shfl_xor(v, m, 32));
      float mnew  = fmaxf(mrow[r], v);
      float alpha = __expf(mrow[r] - mnew);
      mrow[r] = mnew;
      lrow[r] *= alpha;
      #pragma unroll
      for (int t = 0; t < 4; ++t) o[t][r] *= alpha;
    }

    // ---- P = exp(S - m), row sums, stage P to LDS in f16 ----
    float lsum[8];
    #pragma unroll
    for (int r = 0; r < 8; ++r) lsum[r] = 0.f;
    #pragma unroll
    for (int t = 0; t < 4; ++t) {
      #pragma unroll
      for (int r = 0; r < 8; ++r) {
        float p = __expf(s[t][r] - mrow[r]);
        lsum[r] += p;
        pl[(r + 8 * hi) * 72 + t * 16 + m16] = (_Float16)p;
      }
    }
    #pragma unroll
    for (int r = 0; r < 8; ++r) {
      float v = lsum[r];
      #pragma unroll
      for (int m = 1; m < 16; m <<= 1) v += __shfl_xor(v, m, 32);
      lrow[r] += v;
    }
    __syncthreads();

    // ---- O += P * V  (P and V fragments from LDS) ----
    #pragma unroll
    for (int kc = 0; kc < 2; ++kc) {
      const _Float16* pr = pl + (size_t)m16 * 72 + kc * 32 + hi * 16;
      v16h pf = pack16(*(const v8h*)pr, *(const v8h*)(pr + 8));
      #pragma unroll
      for (int t = 0; t < 4; ++t) {
        const _Float16* vr = vl + (size_t)(t * 16 + m16) * 72 + kc * 32 + hi * 16;
        v16h vf = pack16(*(const v8h*)vr, *(const v8h*)(vr + 8));
        o[t] = __builtin_amdgcn_wmma_f32_16x16x32_f16(
            false, pf, false, vf, (short)0, o[t], false, false);
      }
    }
    __syncthreads();   // all waves done reading kl/vl before next staging
  }

  // ---- normalize + store to attn-out [b, n, h*64+d] in f16 ----
  _Float16* outp = Aout + ((size_t)b * SEQ + q0) * CDIM + h * DHEAD;
  #pragma unroll
  for (int r = 0; r < 8; ++r) {
    float inv = 1.f / lrow[r];
    int rw = r + 8 * hi;
    #pragma unroll
    for (int t = 0; t < 4; ++t)
      outp[(size_t)rw * CDIM + t * 16 + m16] = (_Float16)(o[t][r] * inv);
  }
}

// ---------------------------------------------------------------------------
// Output projection + bias + residual -> f32.  grid = (8, 8, 4=batch).
// Same 32x64-per-wave register tiling as the QKV GEMM.
// ---------------------------------------------------------------------------
__global__ void __launch_bounds__(256) out_gemm_kernel(
    const _Float16* __restrict__ Ao, const _Float16* __restrict__ Wto,
    const float* __restrict__ bo, const float* __restrict__ xres,
    float* __restrict__ y)
{
  int w    = threadIdx.x >> 5;
  int lane = threadIdx.x & 31;
  int m16  = lane & 15;
  int hi   = lane >> 4;
  int b    = blockIdx.z;
  int row0 = blockIdx.x * 256 + w * 32;
  int col0 = blockIdx.y * 64;

  const _Float16* A = Ao + (size_t)b * SEQ * CDIM;
  v8f acc[2][4] = {};
  const _Float16* arow0 = A + (size_t)(row0 + m16) * CDIM;
  const _Float16* arow1 = A + (size_t)(row0 + 16 + m16) * CDIM;

  for (int k = 0; k < CDIM; k += 32) {
    int k0 = k + hi * 8;
    v16h a0 = pack16(*(const v8h*)(arow0 + k0), *(const v8h*)(arow0 + k0 + 16));
    v16h a1 = pack16(*(const v8h*)(arow1 + k0), *(const v8h*)(arow1 + k0 + 16));
    #pragma unroll
    for (int t = 0; t < 4; ++t) {
      const _Float16* bp = Wto + (size_t)(col0 + t * 16 + m16) * CDIM + k + hi * 8;
      v16h bfrag = pack16(*(const v8h*)bp, *(const v8h*)(bp + 16));
      acc[0][t] = __builtin_amdgcn_wmma_f32_16x16x32_f16(
          false, a0, false, bfrag, (short)0, acc[0][t], false, false);
      acc[1][t] = __builtin_amdgcn_wmma_f32_16x16x32_f16(
          false, a1, false, bfrag, (short)0, acc[1][t], false, false);
    }
  }

  #pragma unroll
  for (int g = 0; g < 2; ++g) {
    #pragma unroll
    for (int t = 0; t < 4; ++t) {
      int c = col0 + t * 16 + m16;
      float bval = bo[c];
      #pragma unroll
      for (int r = 0; r < 8; ++r) {
        size_t idx = ((size_t)b * SEQ + row0 + g * 16 + r + 8 * hi) * CDIM + c;
        y[idx] = acc[g][t][r] + bval + xres[idx];
      }
    }
  }
}

// ---------------------------------------------------------------------------
extern "C" void kernel_launch(void* const* d_in, const int* in_sizes, int n_in,
                              void* d_out, int out_size, void* d_ws, size_t ws_size,
                              hipStream_t stream) {
  const float* x    = (const float*)d_in[0];
  const float* ctx  = (const float*)d_in[1];
  const float* g_q  = (const float*)d_in[2];
  const float* b_q  = (const float*)d_in[3];
  const float* g_kv = (const float*)d_in[4];
  const float* b_kv = (const float*)d_in[5];
  const float* Wq   = (const float*)d_in[6];
  const float* bq   = (const float*)d_in[7];
  const float* Wk   = (const float*)d_in[8];
  const float* bk   = (const float*)d_in[9];
  const float* Wv   = (const float*)d_in[10];
  const float* bv   = (const float*)d_in[11];
  const float* Wo   = (const float*)d_in[12];
  const float* bo   = (const float*)d_in[13];
  float* out = (float*)d_out;

  const size_t ACT = (size_t)4 * SEQ * CDIM * sizeof(_Float16);  // 8 MiB
  char* ws = (char*)d_ws;
  _Float16* xn = (_Float16*)(ws);
  _Float16* cn = (_Float16*)(ws + ACT);
  _Float16* Wt = (_Float16*)(ws + 2 * ACT);                      // 4 matrices, 2 MiB
  _Float16* Qb = (_Float16*)(ws + 2 * ACT + 4 * CDIM * CDIM * sizeof(_Float16));
  _Float16* Kb = (_Float16*)((char*)Qb + ACT);
  _Float16* Vt = (_Float16*)((char*)Kb + ACT);
  _Float16* AO = (_Float16*)((char*)Vt + ACT);

  // 1) LayerNorms -> f16
  ln_f16_kernel<<<4 * SEQ, 128, 0, stream>>>(x,   g_q,  b_q,  xn, CDIM);
  ln_f16_kernel<<<4 * SEQ, 128, 0, stream>>>(ctx, g_kv, b_kv, cn, CDIM);

  // 2) Weight transpose/convert (q,k,v,o)
  wconv_kernel<<<dim3(1024, 4), 256, 0, stream>>>(Wq, Wk, Wv, Wo, Wt);

  // 3) Q/K/V projections (z = b*3 + proj)
  qkv_gemm_kernel<<<dim3(8, 8, 12), 256, 0, stream>>>(
      xn, cn, Wt, bq, bk, bv, Qb, Kb, Vt);

  // 4) Attention
  attn_kernel<<<dim3(SEQ / 64, 4 * HEADS), 128, 0, stream>>>(Qb, Kb, Vt, AO);

  // 5) Output projection + residual
  out_gemm_kernel<<<dim3(8, 8, 4), 256, 0, stream>>>(
      AO, Wt + (size_t)3 * CDIM * CDIM, bo, x, out);
}
